// LigerLMHeadJSD_18245021073711
// MI455X (gfx1250) — matrix operand order repaired and checked by reading, not compiled, not run on hardware
//
#include <hip/hip_runtime.h>
#include <hip/hip_bf16.h>

// ---------------------------------------------------------------------------
// Fused LM-head generalized JSD (Liger semantics), MI455X / gfx1250.
//
// Roofline: weights = 1.05 GB read (dominant) @ 23.3 TB/s ~= 45us; FLOPs
// (1.07 TF f32, x3 via bf16 hi/lo emulation) are negligible vs HBM time.
// => stream weights once, WMMA bf16x3 for f32-accurate matmul, logits staged
// in d_ws, reductions ride L2.
// ---------------------------------------------------------------------------

typedef __attribute__((ext_vector_type(16))) __bf16 v16bf;
typedef __attribute__((ext_vector_type(8)))  float  v8f;

static constexpr int BT = 2048;
static constexpr int H  = 4096;
static constexpr int V  = 32000;
static constexpr float BETA = 0.5f;

__device__ __forceinline__ unsigned f2u(float x) { union { float f; unsigned u; } c; c.f = x; return c.u; }
__device__ __forceinline__ float u2f(unsigned x) { union { float f; unsigned u; } c; c.u = x; return c.f; }

// Split 16 f32 into bf16 hi (truncation) + bf16 lo (residual), packed in the
// WMMA element order (element 2j in low 16 bits of dword j, 2j+1 in high).
// Pure integer ops: no dependence on __bf16 scalar conversion support.
__device__ __forceinline__ void split16(const float* v, v16bf& hi, v16bf& lo) {
    union { v16bf bf; unsigned u[8]; } Hh, Ll;
#pragma unroll
    for (int j = 0; j < 8; ++j) {
        float a0 = v[2 * j], a1 = v[2 * j + 1];
        unsigned h0 = f2u(a0) & 0xFFFF0000u;
        unsigned h1 = f2u(a1) & 0xFFFF0000u;
        float r0 = a0 - u2f(h0);           // exact (truncation residual)
        float r1 = a1 - u2f(h1);
        unsigned l0 = f2u(r0) & 0xFFFF0000u;
        unsigned l1 = f2u(r1) & 0xFFFF0000u;
        Hh.u[j] = (h0 >> 16) | h1;
        Ll.u[j] = (l0 >> 16) | l1;
    }
    hi = Hh.bf;
    lo = Ll.bf;
}

// A-matrix 16x32 bf16 layout: lane (m, half hh): elements 0..7 <-> K = k0+hh*8+e,
// elements 8..15 <-> K = k0+16+hh*8+(e-8).  p points at row + k0 + hh*8.
__device__ __forceinline__ void load_split_A(const float* __restrict__ p, v16bf& hi, v16bf& lo) {
    float v[16];
    *(float4*)(&v[0])  = *(const float4*)(p + 0);
    *(float4*)(&v[4])  = *(const float4*)(p + 4);
    *(float4*)(&v[8])  = *(const float4*)(p + 16);
    *(float4*)(&v[12]) = *(const float4*)(p + 20);
    split16(v, hi, lo);
}

// B-matrix 32x16 bf16 layout: lane (n, half hh): elements 0..7 <-> K = k0+hh*8+e,
// elements 8..15 <-> K = k0+16+hh*8+(e-8).  p points at row + k0 + hh*8.
__device__ __forceinline__ void load_split_B(const float* __restrict__ p, v16bf& hi, v16bf& lo) {
    float v[16];
    *(float4*)(&v[0])  = *(const float4*)(p + 0);
    *(float4*)(&v[4])  = *(const float4*)(p + 4);
    *(float4*)(&v[8])  = *(const float4*)(p + 16);
    *(float4*)(&v[12]) = *(const float4*)(p + 20);
    split16(v, hi, lo);
}

__device__ __forceinline__ v8f wmma_bf16(const v16bf& a, const v16bf& b, const v8f& c) {
    return __builtin_amdgcn_wmma_f32_16x16x32_bf16(
        /*neg_a=*/false, a, /*neg_b=*/false, b,
        /*c_mod=*/(short)0, c, /*reuse_a=*/false, /*reuse_b=*/false);
}

// ---------------------------------------------------------------------------
// Kernel 1: logits GEMM.  grid = (BT/16, V/128, 2 [student|teacher]),
// block = 256 (8 waves); each wave owns one 16x16 logits tile.
// blockIdx.x (token tiles) varies fastest -> the 128 workgroups sharing a
// weight tile run concurrently -> weights pulled from HBM ~once via L2.
// ---------------------------------------------------------------------------
__global__ void __launch_bounds__(256)
jsd_logits_gemm(const float* __restrict__ Xs, const float* __restrict__ Xt,
                const float* __restrict__ Ws, const float* __restrict__ Wt,
                float* __restrict__ Ls, float* __restrict__ Lp) {
    const int wave = threadIdx.x >> 5;
    const int lane = threadIdx.x & 31;
    const int ml   = lane & 15;   // row-within-tile for A, col for B/C
    const int hh   = lane >> 4;   // lane half

    const int tokenBase = blockIdx.x * 16;
    const int vocabBase = blockIdx.y * 128 + wave * 16;

    const float* __restrict__ X = blockIdx.z ? Xt : Xs;
    const float* __restrict__ W = blockIdx.z ? Wt : Ws;
    float* __restrict__ L       = blockIdx.z ? Lp : Ls;

    const float* __restrict__ arow = X + (size_t)(tokenBase + ml) * H;
    const float* __restrict__ brow = W + (size_t)(vocabBase + ml) * H;

    v8f acc = {};
    for (int k0 = 0; k0 < H; k0 += 32) {
        v16bf ahi, alo, bhi, blo;
        load_split_A(arow + k0 + hh * 8,  ahi, alo);
        load_split_B(brow + k0 + hh * 8,  bhi, blo);
        // f32-accurate product: hi*hi + hi*lo + lo*hi (lo*lo ~ 2^-32, dropped)
        acc = wmma_bf16(ahi, bhi, acc);
        acc = wmma_bf16(ahi, blo, acc);
        acc = wmma_bf16(alo, bhi, acc);
    }

    // C layout: VGPR r -> M = r + hh*8, N = lane&15.  (TEMPERATURE == 1.0)
    float* __restrict__ out = L + (size_t)(tokenBase + hh * 8) * V + vocabBase + ml;
#pragma unroll
    for (int r = 0; r < 8; ++r)
        out[(size_t)r * V] = acc[r];
}

// ---------------------------------------------------------------------------
// Kernel 2: per-token logsumexp (both distributions, single online pass) +
// generalized JSD contribution.  One 256-thread workgroup per token; the
// 2x128KB row stays in L2 between the two sweeps.
// ---------------------------------------------------------------------------
__global__ void __launch_bounds__(256)
jsd_token_loss(const float* __restrict__ Ls, const float* __restrict__ Lp,
               float* __restrict__ tokLoss) {
    __shared__ float shm[256];
    __shared__ float shs[256];
    __shared__ float bcast[2];

    const int t   = blockIdx.x;
    const int tid = threadIdx.x;
    const float* __restrict__ ls = Ls + (size_t)t * V;
    const float* __restrict__ lp = Lp + (size_t)t * V;

    // ---- online logsumexp for student (q) and teacher (p) rows ----
    float mq = -3.0e38f, sq = 0.f;
    float mp = -3.0e38f, sp = 0.f;
    for (int i = tid; i < V; i += 256) {
        float x = ls[i];
        float mn = fmaxf(mq, x);
        sq = sq * __expf(mq - mn) + __expf(x - mn);
        mq = mn;
        float y = lp[i];
        mn = fmaxf(mp, y);
        sp = sp * __expf(mp - mn) + __expf(y - mn);
        mp = mn;
    }

    float lse[2];
#pragma unroll
    for (int which = 0; which < 2; ++which) {
        float m = which ? mp : mq;
        float s = which ? sp : sq;
        shm[tid] = m;
        shs[tid] = s;
        __syncthreads();
        for (int off = 128; off > 0; off >>= 1) {
            if (tid < off) {
                float m1 = shm[tid], s1 = shs[tid];
                float m2 = shm[tid + off], s2 = shs[tid + off];
                float mm = fmaxf(m1, m2);
                shm[tid] = mm;
                shs[tid] = s1 * __expf(m1 - mm) + s2 * __expf(m2 - mm);
            }
            __syncthreads();
        }
        if (tid == 0) bcast[which] = shm[0] + __logf(shs[0]);
        __syncthreads();
        lse[which] = bcast[which];
        __syncthreads();
    }
    const float lseq = lse[0], lsep = lse[1];

    // ---- JSD accumulation:
    //   log_m = logaddexp(log_p + log(beta), log_q + log(1-beta))
    //   loss += beta*p*(log_p-log_m) + (1-beta)*q*(log_q-log_m)
    const float logB  = __logf(BETA);
    const float log1B = __logf(1.0f - BETA);
    float acc = 0.f;
    for (int i = tid; i < V; i += 256) {
        float lq = ls[i] - lseq;
        float lpv = lp[i] - lsep;
        float x = lpv + logB;
        float y = lq + log1B;
        float mx = fmaxf(x, y);
        float mnv = fminf(x, y);
        float lm = mx + log1pf(__expf(mnv - mx));
        acc += BETA * __expf(lpv) * (lpv - lm) + (1.0f - BETA) * __expf(lq) * (lq - lm);
    }
    shm[tid] = acc;
    __syncthreads();
    for (int off = 128; off > 0; off >>= 1) {
        if (tid < off) shm[tid] += shm[tid + off];
        __syncthreads();
    }
    if (tid == 0) tokLoss[t] = shm[0] * (1.0f / (float)BT);
}

// ---------------------------------------------------------------------------
// Kernel 3: deterministic final sum of per-token losses (no float atomics so
// graph replays are bit-stable).
// ---------------------------------------------------------------------------
__global__ void __launch_bounds__(256)
jsd_final_sum(const float* __restrict__ tokLoss, float* __restrict__ out) {
    __shared__ float sh[256];
    const int tid = threadIdx.x;
    float acc = 0.f;
    for (int i = tid; i < BT; i += 256) acc += tokLoss[i];
    sh[tid] = acc;
    __syncthreads();
    for (int off = 128; off > 0; off >>= 1) {
        if (tid < off) sh[tid] += sh[tid + off];
        __syncthreads();
    }
    if (tid == 0) out[0] = sh[0];
}

extern "C" void kernel_launch(void* const* d_in, const int* in_sizes, int n_in,
                              void* d_out, int out_size, void* d_ws, size_t ws_size,
                              hipStream_t stream) {
    (void)in_sizes; (void)n_in; (void)out_size; (void)ws_size;

    const float* Xs = (const float*)d_in[0];  // student_input  (BT,H)
    const float* Xt = (const float*)d_in[1];  // teacher_input  (BT,H)
    const float* Ws = (const float*)d_in[2];  // student_weight (V,H)
    const float* Wt = (const float*)d_in[3];  // teacher_weight (V,H)
    float* out = (float*)d_out;

    // Workspace: logits_s | logits_p | per-token loss
    float* logits_s = (float*)d_ws;
    float* logits_p = logits_s + (size_t)BT * V;
    float* tokLoss  = logits_p + (size_t)BT * V;

    dim3 gemmGrid(BT / 16, V / 128, 2);   // token tiles fastest -> L2 weight reuse
    jsd_logits_gemm<<<gemmGrid, 256, 0, stream>>>(Xs, Xt, Ws, Wt, logits_s, logits_p);
    jsd_token_loss<<<BT, 256, 0, stream>>>(logits_s, logits_p, tokLoss);
    jsd_final_sum<<<1, 256, 0, stream>>>(tokLoss, out);
}
